// MultiHeadAttention_53223234732409
// MI455X (gfx1250) — matrix operand order repaired and checked by reading, not compiled, not run on hardware
//
#include <hip/hip_runtime.h>
#include <hip/hip_bf16.h>

#define DM     1024
#define S_LEN  2048
#define NHEAD  16
#define DHEAD  64
#define BATCH  4
#define NEGVAL (-1000000000.0f)

typedef __attribute__((ext_vector_type(16))) __bf16 bf16x16;
typedef __attribute__((ext_vector_type(8)))  float  f32x8;

union FragU { unsigned u[8]; bf16x16 v; };

__device__ __forceinline__ unsigned short f32_to_bf16(float f) {
  union { float f; unsigned u; } x; x.f = f;
  unsigned r = x.u + 0x7FFFu + ((x.u >> 16) & 1u);
  return (unsigned short)(r >> 16);
}

// 16-byte async DMA global -> LDS (gfx1250, ASYNCcnt-tracked, bypasses VGPRs)
__device__ __forceinline__ void async_copy16(const void* gsrc, unsigned lds_off) {
  unsigned long long ga = (unsigned long long)gsrc;
  asm volatile("global_load_async_to_lds_b128 %0, %1, off"
               :: "v"(lds_off), "v"(ga) : "memory");
}
__device__ __forceinline__ void wait_async0() {
  asm volatile("s_wait_asynccnt 0x0" ::: "memory");
}
__device__ __forceinline__ unsigned lds_off32(const void* p) {
  return (unsigned)(unsigned long long)p;   // low 32 bits = LDS byte offset
}

// ---- WMMA fragment gathers, per CDNA5 ISA 7.12.2 (16-bit A 16x32, B 32x16) ----
__device__ __forceinline__ bf16x16 load_frag_A(const unsigned short* base, int ld) {
  const int lane = threadIdx.x & 31;
  const int m    = lane & 15;
  const int half = lane >> 4;
  FragU r;
#pragma unroll
  for (int v = 0; v < 8; ++v) {
    int k = ((v & 4) ? 16 : 0) + 2 * (v & 3) + half * 8;
    r.u[v] = *(const unsigned*)(base + m * ld + k);
  }
  return r.v;
}
__device__ __forceinline__ bf16x16 load_frag_B(const unsigned short* base, int ld) {
  const int lane = threadIdx.x & 31;
  const int n  = lane & 15;
  const int kh = lane >> 4;
  FragU r;
#pragma unroll
  for (int v = 0; v < 8; ++v) {
    int k = ((v & 4) ? 16 : 0) + 2 * (v & 3) + kh * 8;
    r.u[v] = *(const unsigned*)(base + n * ld + k);
  }
  return r.v;
}

__device__ __forceinline__ f32x8 wmma_bf16(bf16x16 a, bf16x16 b, f32x8 c) {
  return __builtin_amdgcn_wmma_f32_16x16x32_bf16(
      false, a, false, b, (short)0, c, false, false);
}

// --------------------------------------------------------------------------
__global__ void cvt_bf16_kernel(const float* __restrict__ src,
                                unsigned short* __restrict__ dst, int n) {
  int i = blockIdx.x * blockDim.x + threadIdx.x;
  int stride = gridDim.x * blockDim.x;
  for (; i < n; i += stride) dst[i] = f32_to_bf16(src[i]);
}

// --------------------------------------------------------------------------
// C[M,N] = A[M,DM](bf16) @ W[N,DM]^T(bf16).  Block tile 128x64, 8 waves,
// double-buffered LDS tiles with async-DMA / WMMA overlap.
// modes: 0: bf16 [B,H,S,64]; 1: bf16 [B,H,64,S]; 2: f32 flat [M,DM]
__global__ __launch_bounds__(256)
void proj_gemm_kernel(const unsigned short* __restrict__ A,
                      const unsigned short* __restrict__ W,
                      unsigned short* __restrict__ outb,
                      float* __restrict__ outf,
                      int mode) {
  const int LDA = 40, LDB = 40;
  __shared__ unsigned short sA[2][128 * 40];
  __shared__ unsigned short sB[2][64 * 40];

  const int tid  = threadIdx.x;
  const int wave = tid >> 5;
  const int lane = tid & 31;
  const int m0 = blockIdx.x * 128;
  const int n0 = blockIdx.y * 64;

  auto stage = [&](int buf, int k0) {
#pragma unroll
    for (int i = 0; i < 2; ++i) {
      int c = tid + i * 256;
      int row = c >> 2, kc = (c & 3) << 3;
      async_copy16(A + (size_t)(m0 + row) * DM + k0 + kc,
                   lds_off32(&sA[buf][row * LDA + kc]));
    }
    int row = tid >> 2, kc = (tid & 3) << 3;
    async_copy16(W + (size_t)(n0 + row) * DM + k0 + kc,
                 lds_off32(&sB[buf][row * LDB + kc]));
  };

  f32x8 acc[4];
#pragma unroll
  for (int j = 0; j < 4; ++j) acc[j] = {};

  stage(0, 0);
  for (int k0 = 0; k0 < DM; k0 += 32) {
    const int cur = (k0 >> 5) & 1;
    wait_async0();         // only cur-buffer DMAs are outstanding here
    __syncthreads();       // cur visible to all waves; safe to overwrite buf^1
    if (k0 + 32 < DM) stage(cur ^ 1, k0 + 32);
    if (k0 + 64 < DM) {
      __builtin_prefetch(A + (size_t)(m0 + (tid >> 1)) * DM + k0 + 64, 0, 1);
      __builtin_prefetch(W + (size_t)(n0 + (tid & 63)) * DM + k0 + 64, 0, 1);
    }

    bf16x16 afrag = load_frag_A(&sA[cur][wave * 16 * LDA], LDA);
#pragma unroll
    for (int j = 0; j < 4; ++j) {
      bf16x16 bfrag = load_frag_B(&sB[cur][j * 16 * LDB], LDB);
      acc[j] = wmma_bf16(afrag, bfrag, acc[j]);
    }
  }

  const int halfOff = (lane >> 4) * 8;
#pragma unroll
  for (int j = 0; j < 4; ++j) {
#pragma unroll
    for (int r = 0; r < 8; ++r) {
      int m = m0 + wave * 16 + r + halfOff;
      int n = n0 + j * 16 + (lane & 15);
      float val = acc[j][r];
      if (mode == 2) {
        outf[(size_t)m * DM + n] = val;
      } else {
        int b = m / S_LEN, s = m % S_LEN;
        int h = n / DHEAD, d = n & (DHEAD - 1);
        size_t bh = (size_t)b * NHEAD + h;
        if (mode == 0)
          outb[(bh * S_LEN + s) * DHEAD + d] = f32_to_bf16(val);
        else
          outb[(bh * DHEAD + d) * S_LEN + s] = f32_to_bf16(val);
      }
    }
  }
}

// --------------------------------------------------------------------------
// Flash-style attention.  Q,K: [B*H, S, 64] bf16; Vt: [B*H, 64, S] bf16.
// Block = 8 waves x 16 q-rows; double-buffered LDS K/V strips shared by all.
__global__ __launch_bounds__(256)
void attn_kernel(const unsigned short* __restrict__ Q,
                 const unsigned short* __restrict__ K,
                 const unsigned short* __restrict__ Vt,
                 const int* __restrict__ mask,
                 unsigned short* __restrict__ O) {
  const int LDP = 34;
  __shared__ unsigned short sK[2][32 * 64];   // [key_local][d]
  __shared__ unsigned short sV[2][64 * 32];   // [d][key_local]
  __shared__ unsigned short sP[8][16 * 34];

  const int tid  = threadIdx.x;
  const int wave = tid >> 5;
  const int lane = tid & 31;
  const int bh = blockIdx.x;
  const int b  = bh / NHEAD;
  const int h  = bh % NHEAD;
  const int q0 = blockIdx.y * 128 + wave * 16;

  const unsigned short* Qh = Q  + (size_t)bh * S_LEN * DHEAD;
  const unsigned short* Kh = K  + (size_t)bh * S_LEN * DHEAD;
  const unsigned short* Vh = Vt + (size_t)bh * DHEAD * S_LEN;
  const int* mrow_b = mask + (size_t)b * S_LEN;

  auto stageKV = [&](int buf, int kt) {
    int krow = tid >> 3, kkc = (tid & 7) << 3;
    async_copy16(Kh + (size_t)(kt + krow) * DHEAD + kkc,
                 lds_off32(&sK[buf][krow * 64 + kkc]));
    int drow = tid >> 2, vkc = (tid & 3) << 3;
    async_copy16(Vh + (size_t)drow * S_LEN + kt + vkc,
                 lds_off32(&sV[buf][drow * 32 + vkc]));
  };

  bf16x16 qa0 = load_frag_A(Qh + (size_t)q0 * DHEAD + 0,  DHEAD);
  bf16x16 qa1 = load_frag_A(Qh + (size_t)q0 * DHEAD + 32, DHEAD);

  const int halfOff = (lane >> 4) * 8;
  unsigned vq = 0;
#pragma unroll
  for (int r = 0; r < 8; ++r)
    if (mrow_b[q0 + r + halfOff] == 0) vq |= 1u << r;

  float mrun[8], lrun[8];
  f32x8 o[4];
#pragma unroll
  for (int j = 0; j < 4; ++j) o[j] = {};
#pragma unroll
  for (int r = 0; r < 8; ++r) { mrun[r] = -INFINITY; lrun[r] = 0.0f; }

  stageKV(0, 0);
  for (int kt = 0; kt < S_LEN; kt += 32) {
    const int cur = (kt >> 5) & 1;
    wait_async0();
    __syncthreads();
    if (kt + 32 < S_LEN) stageKV(cur ^ 1, kt + 32);

    // scores: two 16x16 col tiles over this 32-key strip
    f32x8 sc[2];
#pragma unroll
    for (int t = 0; t < 2; ++t) {
      bf16x16 kb0 = load_frag_B(&sK[cur][(t * 16) * 64 + 0],  64);
      bf16x16 kb1 = load_frag_B(&sK[cur][(t * 16) * 64 + 32], 64);
      f32x8 c = {};
      c = wmma_bf16(qa0, kb0, c);
      c = wmma_bf16(qa1, kb1, c);
      sc[t] = c;
    }

    // mask + scale (1/sqrt(64) = 0.125)
#pragma unroll
    for (int t = 0; t < 2; ++t) {
      int key = kt + t * 16 + (lane & 15);
      bool vk = (mrow_b[key] == 0);
#pragma unroll
      for (int r = 0; r < 8; ++r)
        sc[t][r] = (vk && ((vq >> r) & 1u)) ? sc[t][r] * 0.125f : NEGVAL;
    }

    // online softmax; rows live across a 16-lane group
    float corr[8];
#pragma unroll
    for (int r = 0; r < 8; ++r) {
      float v = fmaxf(sc[0][r], sc[1][r]);
#pragma unroll
      for (int off = 1; off < 16; off <<= 1)
        v = fmaxf(v, __shfl_xor(v, off, 16));
      float mnew = fmaxf(mrun[r], v);
      corr[r] = __expf(mrun[r] - mnew);
      mrun[r] = mnew;
    }
#pragma unroll
    for (int t = 0; t < 2; ++t)
#pragma unroll
      for (int r = 0; r < 8; ++r)
        sc[t][r] = __expf(sc[t][r] - mrun[r]);
#pragma unroll
    for (int r = 0; r < 8; ++r) {
      float s = sc[0][r] + sc[1][r];
#pragma unroll
      for (int off = 1; off < 16; off <<= 1)
        s += __shfl_xor(s, off, 16);
      lrun[r] = lrun[r] * corr[r] + s;
    }

    // P (16x32) -> per-wave LDS scratch, regather as A fragment
    unsigned short* pw = &sP[wave][0];
#pragma unroll
    for (int t = 0; t < 2; ++t)
#pragma unroll
      for (int r = 0; r < 8; ++r)
        pw[(r + halfOff) * LDP + t * 16 + (lane & 15)] = f32_to_bf16(sc[t][r]);
    asm volatile("s_wait_dscnt 0x0" ::: "memory");
    bf16x16 pa = load_frag_A(pw, LDP);

    // o = o*corr + P @ V
#pragma unroll
    for (int j = 0; j < 4; ++j) {
#pragma unroll
      for (int r = 0; r < 8; ++r) o[j][r] *= corr[r];
      bf16x16 vb = load_frag_B(&sV[cur][(j * 16) * 32], 32);
      o[j] = wmma_bf16(pa, vb, o[j]);
    }
  }

  // write O[b*S + s][h*64 + d] (bf16) for the final projection
#pragma unroll
  for (int j = 0; j < 4; ++j)
#pragma unroll
    for (int r = 0; r < 8; ++r) {
      int s = q0 + r + halfOff;
      int d = j * 16 + (lane & 15);
      float val = o[j][r] / lrun[r];
      O[((size_t)b * S_LEN + s) * DM + h * DHEAD + d] = f32_to_bf16(val);
    }
}

// --------------------------------------------------------------------------
extern "C" void kernel_launch(void* const* d_in, const int* in_sizes, int n_in,
                              void* d_out, int out_size, void* d_ws, size_t ws_size,
                              hipStream_t stream) {
  const float* x    = (const float*)d_in[0];
  const int*   mask = (const int*)d_in[1];
  const float* Wq   = (const float*)d_in[2];
  const float* Wk   = (const float*)d_in[3];
  const float* Wv   = (const float*)d_in[4];
  const float* Wo   = (const float*)d_in[5];
  float* out = (float*)d_out;

  const size_t M = (size_t)BATCH * S_LEN;   // 8192
  unsigned short* xb  = (unsigned short*)d_ws;
  unsigned short* wqb = xb  + M * DM;
  unsigned short* wkb = wqb + (size_t)DM * DM;
  unsigned short* wvb = wkb + (size_t)DM * DM;
  unsigned short* wob = wvb + (size_t)DM * DM;
  unsigned short* Qb  = wob + (size_t)DM * DM;
  unsigned short* Kb  = Qb  + M * DM;
  unsigned short* Vtb = Kb  + M * DM;
  unsigned short* Ob  = Vtb + M * DM;

  cvt_bf16_kernel<<<2048, 256, 0, stream>>>(x,  xb,  (int)(M * DM));
  cvt_bf16_kernel<<<512,  256, 0, stream>>>(Wq, wqb, DM * DM);
  cvt_bf16_kernel<<<512,  256, 0, stream>>>(Wk, wkb, DM * DM);
  cvt_bf16_kernel<<<512,  256, 0, stream>>>(Wv, wvb, DM * DM);
  cvt_bf16_kernel<<<512,  256, 0, stream>>>(Wo, wob, DM * DM);

  dim3 gp(64, 16);  // (8192/128, 1024/64)
  proj_gemm_kernel<<<gp, 256, 0, stream>>>(xb, wqb, Qb,  nullptr, 0);
  proj_gemm_kernel<<<gp, 256, 0, stream>>>(xb, wkb, Kb,  nullptr, 0);
  proj_gemm_kernel<<<gp, 256, 0, stream>>>(xb, wvb, Vtb, nullptr, 1);

  attn_kernel<<<dim3(BATCH * NHEAD, S_LEN / 128), 256, 0, stream>>>(Qb, Kb, Vtb, mask, Ob);

  proj_gemm_kernel<<<gp, 256, 0, stream>>>(Ob, wob, nullptr, out, 2);
}
